// MNeighborAttent_65910568125068
// MI455X (gfx1250) — matrix operand order repaired and checked by reading, not compiled, not run on hardware
//
#include <hip/hip_runtime.h>

// ---------------------------------------------------------------------------
// MNeighborAttent for MI455X (gfx1250).
//  - All GEMMs: v_wmma_f32_16x16x32_f16 (f16 in, f32 accumulate).
//  - sim_ns factored:  kbar = sum_n ns_w[n]*k[idx[n]]  ->  one fused GEMM.
//  - Neighbor Gram (17 GFLOP, the dominant term): per-block cooperative LDS
//    staging of each 64x256 qn j-block through GLOBAL_LOAD_ASYNC_TO_LDS_B128
//    in a pre-swizzled layout (one contiguous 32B LDS read per lane per WMMA
//    B fragment) -> 4x less L2 traffic than per-wave global reads.
//  - Score rows (16 x T) live in LDS; probs alias the same storage (f16 into
//    f32 slots) so S(32KB) + Bstage(32KB) = exactly 64KB static LDS.
// ---------------------------------------------------------------------------

typedef _Float16 half_t;
typedef __attribute__((ext_vector_type(8)))  _Float16 v8h;
typedef __attribute__((ext_vector_type(16))) _Float16 v16h;
typedef __attribute__((ext_vector_type(8)))  float    v8f;
typedef __attribute__((ext_vector_type(4)))  float    v4f;

constexpr int NB = 8;    // batch
constexpr int NT = 512;  // tokens
constexpr int NC = 256;  // channels
constexpr int NK = 4;    // top-k neighbors

__device__ __forceinline__ int lane_id() { return (int)(threadIdx.x & 31); }

__device__ __forceinline__ unsigned lds_off(const void* p) {
  // addrspacecast(shared->flat) places the LDS offset in the low 32 bits
  return (unsigned)(unsigned long long)p;
}
__device__ __forceinline__ void async_copy_b128(unsigned lds_byte,
                                                const void* gsrc) {
  asm volatile("global_load_async_to_lds_b128 %0, %1, off"
               :: "v"(lds_byte), "v"(gsrc) : "memory");
}
__device__ __forceinline__ void wait_async0() {
  asm volatile("s_wait_asynccnt 0x0" ::: "memory");
}

// ---- WMMA fragment loaders (16x16x32 f16; contiguous k = 16*(lane/16) + s, same for A and B) ------------------
// A (16x32), row-major source, K contiguous:
//   lanes 0-15: row m=l, K0..15; lanes 16-31: row m=l-16, K16..31
__device__ __forceinline__ v16h load_fragA(const half_t* p0, int ld) {
  const int l = lane_id();
  const half_t* p = p0 + (l & 15) * ld + (l >> 4) * 16;
  v8h a = *(const v8h*)(p);
  v8h b = *(const v8h*)(p + 8);
  return __builtin_shufflevector(a, b, 0,1,2,3,4,5,6,7,8,9,10,11,12,13,14,15);
}
// B (32x16) sourced from row-major "B^T" (rows = N, K contiguous):
//   lanes 0-15: col n=l, K0..15; lanes 16-31: col n=l-16, K16..31
__device__ __forceinline__ v16h load_fragB(const half_t* p0, int ld) {
  const int l = lane_id();
  const half_t* p = p0 + (l & 15) * ld + (l >> 4) * 16;
  v8h a = *(const v8h*)(p);
  v8h b = *(const v8h*)(p + 8);
  return __builtin_shufflevector(a, b, 0,1,2,3,4,5,6,7,8,9,10,11,12,13,14,15);
}
__device__ __forceinline__ v8f wmma_f16(const v16h& a, const v16h& b, const v8f& c) {
  return __builtin_amdgcn_wmma_f32_16x16x32_f16(false, a, false, b, (short)0, c,
                                                false, false);
}

// ---- tile stores (D layout: VGPR r -> row r + 8*(lane>=16), col = lane&15) --
__device__ __forceinline__ void store_tile_f32(float* dst, int ld, const v8f& d,
                                               float* stage, bool accum) {
  const int l = lane_id();
  const int col = l & 15, hi = l >> 4;
#pragma unroll
  for (int r = 0; r < 8; ++r) stage[(r + 8 * hi) * 16 + col] = d[r];
  asm volatile("s_wait_dscnt 0" ::: "memory");
  const int rr = l >> 1, c8 = (l & 1) * 8;
  v4f s0 = *(const v4f*)(stage + rr * 16 + c8);
  v4f s1 = *(const v4f*)(stage + rr * 16 + c8 + 4);
  float* dp = dst + rr * ld + c8;
  if (accum) {
    v4f o0 = *(const v4f*)(dp);
    v4f o1 = *(const v4f*)(dp + 4);
    s0 = s0 + o0; s1 = s1 + o1;
  }
  *(v4f*)(dp) = s0;
  *(v4f*)(dp + 4) = s1;
}
__device__ __forceinline__ void store_tile_f16(half_t* dst, int ld, const v8f& d,
                                               half_t* stageh) {
  const int l = lane_id();
  const int col = l & 15, hi = l >> 4;
#pragma unroll
  for (int r = 0; r < 8; ++r) stageh[(r + 8 * hi) * 16 + col] = (half_t)d[r];
  asm volatile("s_wait_dscnt 0" ::: "memory");
  const int rr = l >> 1, c8 = (l & 1) * 8;
  v8h s = *(const v8h*)(stageh + rr * 16 + c8);
  *(v8h*)(dst + rr * ld + c8) = s;
}
// transposed f16 store: element (row=token r', col=chan c) -> dstT[c*ldT + r']
__device__ __forceinline__ void store_tile_f16_T(half_t* dstT, int ldT, const v8f& d) {
  const int l = lane_id();
  const int c = l & 15, hi = l >> 4;
  v8h s;
#pragma unroll
  for (int r = 0; r < 8; ++r) s[r] = (half_t)d[r];
  *(v8h*)(dstT + c * ldT + hi * 8) = s;   // rows r..r+7 are consecutive tokens
}

// ---------------------------------------------------------------------------
__global__ void k_cvt(const float* __restrict__ src, half_t* __restrict__ dst, int n) {
  int i = blockIdx.x * 256 + threadIdx.x;
  if (i < n) dst[i] = (half_t)src[i];
}

// recover idx from one-hot inxs (B*T*K rows of T)
__global__ void k_idx(const float* __restrict__ inxs, int* __restrict__ idx) {
  const int row = blockIdx.x;             // B*T*K
  const int j0 = threadIdx.x * 4;         // 128 threads * 4 = 512
#pragma unroll
  for (int e = 0; e < 4; ++e) {
    int p = j0 + e;
    if (inxs[(size_t)row * NT + p] > 0.5f) idx[row] = p;
  }
}

// fused projections: [q | k | qs | v | mq | mk | mv] = x @ Wcat^T
__global__ void __launch_bounds__(256) k_proj(
    const half_t* __restrict__ xh, const half_t* __restrict__ Wh,
    float* __restrict__ qraw, float* __restrict__ kraw, float* __restrict__ qsraw,
    half_t* __restrict__ vT, half_t* __restrict__ mqh, half_t* __restrict__ mkh,
    half_t* __restrict__ mvT) {
  __shared__ __align__(16) half_t xt[16 * NC];
  __shared__ __align__(16) float  stage[8][16 * 16];
  const int b = blockIdx.x >> 5, it = blockIdx.x & 31, t0 = it * 16;
  const half_t* xp = xh + ((size_t)b * NT + t0) * NC;

  // async-stage the 16x256 f16 x tile into LDS (512 x 16B chunks)
  {
    const unsigned base = lds_off(xt);
#pragma unroll
    for (int i = 0; i < 2; ++i) {
      const int c = threadIdx.x + i * 256;
      async_copy_b128(base + c * 16, xp + c * 8);
    }
    wait_async0();
  }
  __syncthreads();

  const int w = threadIdx.x >> 5;
  v16h A[8];
#pragma unroll
  for (int kk = 0; kk < 8; ++kk) A[kk] = load_fragA(xt + kk * 32, NC);

  for (int ot = w; ot < 112; ot += 8) {           // 7C/16 = 112 output tiles
    if (ot + 8 < 112)
      __builtin_prefetch(Wh + (size_t)(ot + 8) * 16 * NC, 0, 1);
    v8f acc = {};
    const half_t* wb = Wh + (size_t)ot * 16 * NC;
#pragma unroll
    for (int kk = 0; kk < 8; ++kk)
      acc = wmma_f16(A[kk], load_fragB(wb + kk * 32, NC), acc);

    const int o0 = ot * 16;
    const size_t rowoff = ((size_t)b * NT + t0) * NC;
    if (o0 < 256) {
      store_tile_f32(qraw + rowoff + o0, NC, acc, stage[w], false);
    } else if (o0 < 512) {
      store_tile_f32(kraw + rowoff + (o0 - 256), NC, acc, stage[w], false);
    } else if (o0 < 768) {
      store_tile_f32(qsraw + rowoff + (o0 - 512), NC, acc, stage[w], false);
    } else if (o0 < 1024) {
      store_tile_f16_T(vT + ((size_t)b * NC + (o0 - 768)) * NT + t0, NT, acc);
    } else if (o0 < 1280) {
      store_tile_f16(mqh + rowoff + (o0 - 1024), NC, acc, (half_t*)stage[w]);
    } else if (o0 < 1536) {
      store_tile_f16(mkh + rowoff + (o0 - 1280), NC, acc, (half_t*)stage[w]);
    } else {
      store_tile_f16_T(mvT + ((size_t)b * NC + (o0 - 1536)) * NT + t0, NT, acc);
    }
  }
}

// rowwise L2 normalize (matches t / max(||t||, 1e-12)), f32 -> f16
__global__ void __launch_bounds__(256) k_norm(const float* __restrict__ raw,
                                              half_t* __restrict__ outh) {
  __shared__ float red[256];
  const int row = blockIdx.x, c = threadIdx.x;
  const float v = raw[(size_t)row * NC + c];
  red[c] = v * v;
  __syncthreads();
  for (int s = 128; s > 0; s >>= 1) {
    if (c < s) red[c] += red[c + s];
    __syncthreads();
  }
  const float inv = 1.0f / fmaxf(sqrtf(red[0]), 1e-12f);
  outh[(size_t)row * NC + c] = (half_t)(v * inv);
}

// gather qn = q[idx]; kbar = sum_n ns_w[n] * k[idx[n]]
__global__ void __launch_bounds__(256) k_gather(
    const half_t* __restrict__ qh, const half_t* __restrict__ kh,
    const int* __restrict__ idx, const float* __restrict__ ns_w,
    half_t* __restrict__ qnh, half_t* __restrict__ kbarh) {
  const int b = blockIdx.x >> 9, t = blockIdx.x & 511, c = threadIdx.x;
  float acc = 0.0f;
#pragma unroll
  for (int n = 0; n < NK; ++n) {
    const int j = idx[((size_t)b * NT + t) * NK + n];
    const half_t qv = qh[((size_t)b * NT + j) * NC + c];
    qnh[(((size_t)b * NT + t) * NK + n) * NC + c] = qv;
    acc += ns_w[n] * (float)kh[((size_t)b * NT + j) * NC + c];
  }
  kbarh[((size_t)b * NT + t) * NC + c] = (half_t)acc;
}

// fused attention: mode 0 = neighbor path (sim_s + sim_ns + sim_nn + mask),
//                  mode 1 = MHA path (scaled dot, accumulate into out)
__global__ void __launch_bounds__(128) k_attn(
    const half_t* __restrict__ Arow,  // qs | mq   (B,T,C) f16
    const half_t* __restrict__ Brow,  // qs | mk
    const half_t* __restrict__ A2row, // kbar | null
    const half_t* __restrict__ B2row, // k    | null
    const half_t* __restrict__ qnh,   // (B,T*K,C) | null
    const float*  __restrict__ adj,   // (B,T,T)   | null
    const float*  __restrict__ nn_w,  // (K)       | null
    const half_t* __restrict__ vTs,   // vT | mvT  (B,C,T) f16
    float* __restrict__ out, int mode, float scale) {
  // 64KB static LDS total:
  //  S    : 16 x 512 f32 scores; after softmax each row's first 1KB is re-used
  //         as 512 f16 probs, and floats [256..511] of row w serve as wave-w
  //         store staging in phase 4.
  //  Bst  : swizzled 64x256 f16 qn j-block; tile (nt,kk) at (nt*8+kk)*1KB,
  //         lane l's 32B B-fragment contiguous at +l*32.
  __shared__ __align__(16) float  S[16][NT];        // 32 KB
  __shared__ __align__(16) half_t Bst[16 * 1024];   // 32 KB
  const int b = blockIdx.x >> 5, it = blockIdx.x & 31, i0 = it * 16;
  const int w = threadIdx.x >> 5, l = lane_id();

  // ---- phase 1: S = A1 @ B1^T (+ A2 @ B2^T + mask | * scale) ----
  v16h A1[8], A2[8];
#pragma unroll
  for (int kk = 0; kk < 8; ++kk)
    A1[kk] = load_fragA(Arow + ((size_t)b * NT + i0) * NC + kk * 32, NC);
  if (mode == 0) {
#pragma unroll
    for (int kk = 0; kk < 8; ++kk)
      A2[kk] = load_fragA(A2row + ((size_t)b * NT + i0) * NC + kk * 32, NC);
  }
  for (int jt = w; jt < 32; jt += 4) {
    v8f acc = {};
    const half_t* bb = Brow + ((size_t)b * NT + jt * 16) * NC;
#pragma unroll
    for (int kk = 0; kk < 8; ++kk)
      acc = wmma_f16(A1[kk], load_fragB(bb + kk * 32, NC), acc);
    if (mode == 0) {
      const half_t* bb2 = B2row + ((size_t)b * NT + jt * 16) * NC;
#pragma unroll
      for (int kk = 0; kk < 8; ++kk)
        acc = wmma_f16(A2[kk], load_fragB(bb2 + kk * 32, NC), acc);
    }
    const int col = l & 15, hi = l >> 4;
#pragma unroll
    for (int r = 0; r < 8; ++r) {
      const int row = r + 8 * hi;
      float v = acc[r];
      if (mode == 0) {
        const float a = adj[((size_t)b * NT + i0 + row) * NT + jt * 16 + col];
        if (a == 0.0f) v += -1e22f;
      } else {
        v *= scale;
      }
      S[row][jt * 16 + col] = v;
    }
  }
  __syncthreads();

  // ---- phase 2 (mode 0): sim_nn via 64x64 Gram tiles of qn ----
  if (mode == 0) {
    float nw[4] = {nn_w[0], nn_w[1], nn_w[2], nn_w[3]};
    // wave w owns G-rows 16w..16w+15 = tokens i0+4w..i0+4w+3 (x = row & 3)
    const half_t* Ab = qnh + ((size_t)b * NT + i0 + 4 * w) * NK * NC;
    v16h An[8];
#pragma unroll
    for (int kk = 0; kk < 8; ++kk) An[kk] = load_fragA(Ab + kk * 32, NC);
    const int hi = l >> 4;
    const unsigned bbase = lds_off(Bst);
    for (int jt = 0; jt < 32; ++jt) {
      // cooperative async stage: 64 rows x 256 halfs, swizzled so each lane's
      // WMMA B fragment is one contiguous 32B LDS read. 2048 x 16B chunks.
      const half_t* Bb = qnh + ((size_t)b * NT + jt * 16) * NK * NC;
#pragma unroll
      for (int i = 0; i < 16; ++i) {
        const int cl = (int)threadIdx.x + i * 128;  // 0..2047
        const int j = cl >> 5;                      // qn row 0..63
        const int h = (cl & 31) * 8;                // half offset in row
        const int kk = h >> 5, p = h & 31;
        const int hi2 = p >> 4, q = p & 15;
        const int dsthalf = (((j >> 4) * 8 + kk) << 9) +
                            (((j & 15) + (hi2 << 4)) << 4) + q;
        async_copy_b128(bbase + dsthalf * 2, Bb + (size_t)j * NC + h);
      }
      wait_async0();
      __syncthreads();
#pragma unroll
      for (int nt = 0; nt < 4; ++nt) {
        v8f g = {};
#pragma unroll
        for (int kk = 0; kk < 8; ++kk) {
          v16h bf = *(const v16h*)(Bst + ((nt * 8 + kk) << 9) + l * 16);
          g = wmma_f16(An[kk], bf, g);
        }
        // max over y (4 adjacent lanes = neighbor cols), nn_w-sum over x rows
        float s0 = 0.0f, s1 = 0.0f;
#pragma unroll
        for (int r = 0; r < 8; ++r) {
          float m = g[r];
          m = fmaxf(m, __shfl_xor(m, 1, 32));
          m = fmaxf(m, __shfl_xor(m, 2, 32));
          const float wv = nw[r & 3];
          if (r < 4) s0 += wv * m; else s1 += wv * m;
        }
        if ((l & 3) == 0) {
          const int jcol = jt * 16 + nt * 4 + ((l & 15) >> 2);
          const int r0 = 4 * w + (hi ? 2 : 0);
          S[r0][jcol]     += s0;   // token i0+r0   (x-rows 0..3)
          S[r0 + 1][jcol] += s1;   // token i0+r0+1 (x-rows 4..7)
        }
      }
      __syncthreads();   // protect Bst before next jt restage
    }
  }
  __syncthreads();

  // ---- phase 3: rowwise softmax; f16 probs written in place over S ----
  for (int rr = 0; rr < 4; ++rr) {
    const int row = 4 * w + rr;
    float vals[16];
#pragma unroll
    for (int t = 0; t < 16; ++t) vals[t] = S[row][l + 32 * t];
    float mx = -3.4e38f;
#pragma unroll
    for (int t = 0; t < 16; ++t) mx = fmaxf(mx, vals[t]);
#pragma unroll
    for (int off = 16; off > 0; off >>= 1) mx = fmaxf(mx, __shfl_xor(mx, off, 32));
    float sum = 0.0f;
#pragma unroll
    for (int t = 0; t < 16; ++t) { vals[t] = __expf(vals[t] - mx); sum += vals[t]; }
#pragma unroll
    for (int off = 16; off > 0; off >>= 1) sum += __shfl_xor(sum, off, 32);
    const float inv = 1.0f / sum;
    half_t* Pr = (half_t*)&S[row][0];    // all reads above are in registers
#pragma unroll
    for (int t = 0; t < 16; ++t) Pr[l + 32 * t] = (half_t)(vals[t] * inv);
  }
  __syncthreads();

  // ---- phase 4: out(16xC) = P(16xT) @ V(TxC) via vT (C,T) source ----
  const half_t* Pb = (const half_t*)&S[0][0];   // row stride 1024 halfs
  float* stage = (float*)&S[w][256];            // free second half of row w
  float* op = out + ((size_t)b * NT + i0) * NC;
  for (int ct = w; ct < 16; ct += 4) {                // C/16 tiles over 4 waves
    v8f acc = {};
    const half_t* vb = vTs + ((size_t)b * NC + ct * 16) * NT;
#pragma unroll
    for (int kk = 0; kk < 16; ++kk) {
      v16h a = load_fragA(Pb + kk * 32, 1024);
      v16h bf = load_fragB(vb + kk * 32, NT);
      acc = wmma_f16(a, bf, acc);
    }
    store_tile_f32(op + ct * 16, NC, acc, stage, /*accum=*/mode == 1);
  }
}

// ---------------------------------------------------------------------------
extern "C" void kernel_launch(void* const* d_in, const int* in_sizes, int n_in,
                              void* d_out, int out_size, void* d_ws, size_t ws_size,
                              hipStream_t stream) {
  const float* x    = (const float*)d_in[0];
  const float* adj  = (const float*)d_in[1];
  const float* inxs = (const float*)d_in[2];
  const float* Wq   = (const float*)d_in[3];
  const float* Wk   = (const float*)d_in[4];
  const float* Wqs  = (const float*)d_in[5];
  const float* Wv   = (const float*)d_in[6];
  const float* nn_w = (const float*)d_in[7];
  const float* ns_w = (const float*)d_in[8];
  const float* Wqkv = (const float*)d_in[9];
  float* out = (float*)d_out;

  char* p = (char*)d_ws;
  auto carve = [&](size_t bytes) -> char* {
    char* r = p;
    p += (bytes + 255) & ~(size_t)255;
    return r;
  };
  const size_t BTC = (size_t)NB * NT * NC;
  half_t* Wh    = (half_t*)carve((size_t)7 * NC * NC * 2);
  half_t* xhb   = (half_t*)carve(BTC * 2);
  float*  qraw  = (float*) carve(BTC * 4);
  float*  kraw  = (float*) carve(BTC * 4);
  float*  qsraw = (float*) carve(BTC * 4);
  half_t* qh    = (half_t*)carve(BTC * 2);
  half_t* kh    = (half_t*)carve(BTC * 2);
  half_t* qsh   = (half_t*)carve(BTC * 2);
  half_t* kbarh = (half_t*)carve(BTC * 2);
  half_t* mqh   = (half_t*)carve(BTC * 2);
  half_t* mkh   = (half_t*)carve(BTC * 2);
  half_t* vT    = (half_t*)carve(BTC * 2);
  half_t* mvT   = (half_t*)carve(BTC * 2);
  half_t* qnh   = (half_t*)carve(BTC * NK * 2);
  int*    idx   = (int*)   carve((size_t)NB * NT * NK * 4);

  const int CC = NC * NC;
  k_cvt<<<(CC + 255) / 256, 256, 0, stream>>>(Wq,   Wh,            CC);
  k_cvt<<<(CC + 255) / 256, 256, 0, stream>>>(Wk,   Wh + CC,       CC);
  k_cvt<<<(CC + 255) / 256, 256, 0, stream>>>(Wqs,  Wh + 2 * CC,   CC);
  k_cvt<<<(CC + 255) / 256, 256, 0, stream>>>(Wv,   Wh + 3 * CC,   CC);
  k_cvt<<<(3 * CC + 255) / 256, 256, 0, stream>>>(Wqkv, Wh + 4 * CC, 3 * CC);
  k_cvt<<<(int)((BTC + 255) / 256), 256, 0, stream>>>(x, xhb, (int)BTC);

  k_idx<<<NB * NT * NK, 128, 0, stream>>>(inxs, idx);

  k_proj<<<NB * (NT / 16), 256, 0, stream>>>(xhb, Wh, qraw, kraw, qsraw,
                                             vT, mqh, mkh, mvT);

  k_norm<<<NB * NT, 256, 0, stream>>>(qraw,  qh);
  k_norm<<<NB * NT, 256, 0, stream>>>(kraw,  kh);
  k_norm<<<NB * NT, 256, 0, stream>>>(qsraw, qsh);

  k_gather<<<NB * NT, 256, 0, stream>>>(qh, kh, idx, ns_w, qnh, kbarh);

  // neighbor-attention path (writes out)
  k_attn<<<NB * (NT / 16), 128, 0, stream>>>(qsh, qsh, kbarh, kh, qnh, adj,
                                             nn_w, vT, out, 0, 1.0f);
  // single-head MHA path (accumulates into out); scale = C^-0.5
  k_attn<<<NB * (NT / 16), 128, 0, stream>>>(mqh, mkh, nullptr, nullptr, nullptr,
                                             nullptr, nullptr, mvT, out, 1, 0.0625f);
}
